// MultiHeadAttention_66048007078442
// MI455X (gfx1250) — matrix operand order repaired and checked by reading, not compiled, not run on hardware
//
#include <hip/hip_runtime.h>
#include <hip/hip_bf16.h>

#define B_ 4
#define S_ 2048
#define D_ 1024
#define H_ 16
#define DK_ 64
#define DV_ 64

typedef __attribute__((ext_vector_type(16))) __bf16 v16bf;
typedef __attribute__((ext_vector_type(8)))  float  v8f;

union FragAB { v16bf v; float4 q4[2]; __bf16 e[16]; };

// A-matrix 16x32 bf16 fragment from row-major [rows x ld] bf16 buffer.
// ISA layout: lane m=L&15; lanes<16 hold K{0..7,16..23}, lanes>=16 hold K{8..15,24..31}.
__device__ inline v16bf loadA(const __bf16* base, int ld, int k0) {
    int lane = threadIdx.x & 31;
    int m = lane & 15, hi = lane >> 4;
    const __bf16* p = base + (size_t)m * ld + k0 + hi * 8;
    FragAB f;
    f.q4[0] = *(const float4*)(p);        // K chunk 0..7 (+hi*8)
    f.q4[1] = *(const float4*)(p + 16);   // K chunk 16..23 (+hi*8)
    return f.v;
}

// B-matrix 32x16 bf16 fragment. base points at (n=0,k=k0) of a panel whose
// 16 rows are the N dimension, each row ld elements of contiguous K.
// ISA layout: lane n=L&15; lanes<16 hold K{0..7,16..23}, lanes>=16 hold K{8..15,24..31}.
__device__ inline v16bf loadB(const __bf16* base, int ld) {
    int lane = threadIdx.x & 31;
    int n = lane & 15, hi = lane >> 4;
    const __bf16* p = base + (size_t)n * ld + hi * 8;
    FragAB f;
    f.q4[0] = *(const float4*)(p);
    f.q4[1] = *(const float4*)(p + 16);
    return f.v;
}

__device__ inline v8f wmma_bf16(v16bf a, v16bf b, v8f c) {
    return __builtin_amdgcn_wmma_f32_16x16x32_bf16(
        false, a, false, b, (short)0, c, false, false);
}

__device__ inline float rowmax16(float v) {
    for (int m = 1; m < 16; m <<= 1) v = fmaxf(v, __shfl_xor(v, m, 32));
    return v;
}
__device__ inline float rowsum16(float v) {
    for (int m = 1; m < 16; m <<= 1) v += __shfl_xor(v, m, 32);
    return v;
}
__device__ inline float allsum32(float v) {
    for (int m = 1; m < 32; m <<= 1) v += __shfl_xor(v, m, 32);
    return v;
}

// ---------------- Kernel 0: weights -> transposed bf16  WT[c][k] = W[k][c]
__global__ __launch_bounds__(256) void cvt_weights_k(
    const float* Wq, const float* Wk, const float* Wv, const float* Wo,
    __bf16* WTq, __bf16* WTk, __bf16* WTv, __bf16* WTo)
{
    size_t idx = (size_t)blockIdx.x * 256 + threadIdx.x;   // 4 * 1M
    int mat = (int)(idx >> 20);
    size_t r = idx & 1048575u;
    int c = (int)(r >> 10);
    int k = (int)(r & 1023u);
    const float* W = (mat == 0) ? Wq : (mat == 1) ? Wk : (mat == 2) ? Wv : Wo;
    __bf16*     WT = (mat == 0) ? WTq : (mat == 1) ? WTk : (mat == 2) ? WTv : WTo;
    WT[((size_t)c << 10) + k] = (__bf16)W[((size_t)k << 10) + c];
}

// ---------------- Kernel 1: fused QKV projection (bf16 WMMA, fp32->bf16 A staged in LDS)
// grid (512, 3): x = 16-row tile, y = which of {q,k,v}. 256 threads = 8 waves.
__global__ __launch_bounds__(256) void qkv_proj_k(
    const float* q, const float* k, const float* v,
    const __bf16* WTq, const __bf16* WTk, const __bf16* WTv,
    __bf16* qh, __bf16* kh, __bf16* vT)
{
    extern __shared__ __bf16 As[];           // 16 x 1024 bf16 = 32 KB
    int which = blockIdx.y;
    const float*   X  = (which == 0) ? q : (which == 1) ? k : v;
    const __bf16*  WT = (which == 0) ? WTq : (which == 1) ? WTk : WTv;
    int r0 = blockIdx.x * 16;

    for (int t = threadIdx.x; t < 16 * 1024; t += 256) {
        int m = t >> 10, c = t & 1023;
        As[m * 1024 + c] = (__bf16)X[(size_t)(r0 + m) * 1024 + c];
    }
    __syncthreads();

    int w = threadIdx.x >> 5, lane = threadIdx.x & 31;
    int n = lane & 15, hi = lane >> 4;

    for (int i = 0; i < 8; i++) {
        int c0 = (w * 8 + i) * 16;
        v8f acc = {};
        const __bf16* Bb = WT + (size_t)c0 * 1024;
        for (int k0 = 0; k0 < 1024; k0 += 32)
            acc = wmma_bf16(loadA(As, 1024, k0), loadB(Bb + k0, 1024), acc);

        for (int g = 0; g < 8; g++) {
            int m = g + hi * 8;
            int r = r0 + m;
            int b_ = r >> 11, s = r & 2047;
            int c = c0 + n;
            int h = c >> 6, dd = c & 63;
            __bf16 val = (__bf16)acc[g];
            if (which == 0)
                qh[(((size_t)b_ * H_ + h) * S_ + s) * DK_ + dd] = val;
            else if (which == 1)
                kh[(((size_t)b_ * H_ + h) * S_ + s) * DK_ + dd] = val;
            else // V stored transposed: vT[b][h][dv][s]
                vT[(((size_t)b_ * H_ + h) * DV_ + dd) * S_ + s] = val;
        }
    }
}

// ---------------- Kernel 2: attention. One block = (b, h, 16 q-rows) x all 2048 keys.
// 128 threads = 4 waves. Exact 3-pass softmax, full fp32 score stripe in LDS (CDNA5 320KB/WGP).
__global__ __launch_bounds__(128) void attention_k(
    const __bf16* qh, const __bf16* kh, const __bf16* vT,
    const unsigned char* mask, float* attn, __bf16* o)
{
    extern __shared__ char smem_raw[];
    __bf16* Qs  = (__bf16*)smem_raw;                                 // 16x64 bf16 (2 KB)
    float*  Sc  = (float*)(smem_raw + 2048);                         // 16x2048 f32 (128 KB)
    __bf16* Ps  = (__bf16*)(smem_raw + 2048 + 131072);               // 16x2048 bf16 (64 KB)
    float*  Red = (float*)(smem_raw + 2048 + 131072 + 65536);        // 4x16 f32

    int bid = blockIdx.x;
    int qt = bid & 127;
    int h  = (bid >> 7) & 15;
    int b  = bid >> 11;

    const __bf16* qbh = qh + (((size_t)b * H_ + h) * S_) * DK_;
    const __bf16* kbh = kh + (((size_t)b * H_ + h) * S_) * DK_;
    const __bf16* vbh = vT + (((size_t)b * H_ + h) * DV_) * S_;
    float* attn_bh = attn + (((size_t)b * H_ + h) * S_ + (size_t)qt * 16) * S_;
    const unsigned char* mrow = mask + ((size_t)b * S_ + (size_t)qt * 16) * S_;

    int w = threadIdx.x >> 5, lane = threadIdx.x & 31;
    int n = lane & 15, hi = lane >> 4;

    for (int t = threadIdx.x; t < 16 * 64; t += 128)
        Qs[t] = qbh[(size_t)(qt * 16 + (t >> 6)) * DK_ + (t & 63)];
    __syncthreads();

    // Pass 1: scores = (Q Kt)/8, mask, stash fp32 in LDS, track row max
    float fm[8];
    for (int g = 0; g < 8; g++) fm[g] = -3.0e38f;
    for (int ct = w; ct < 128; ct += 4) {
        v8f acc = {};
        const __bf16* Bb = kbh + (size_t)ct * 16 * DK_;
        for (int k0 = 0; k0 < 64; k0 += 32)
            acc = wmma_bf16(loadA(Qs, 64, k0), loadB(Bb + k0, 64), acc);
        int c = ct * 16 + n;
        for (int g = 0; g < 8; g++) {
            int m = g + hi * 8;
            float s = acc[g] * 0.125f;
            if (!mrow[(size_t)m * S_ + c]) s = -1.0e9f;
            Sc[m * 2048 + c] = s;
            fm[g] = fmaxf(fm[g], s);
        }
    }
    for (int g = 0; g < 8; g++) fm[g] = rowmax16(fm[g]);
    if (n == 0)
        for (int g = 0; g < 8; g++) Red[w * 16 + g + hi * 8] = fm[g];
    __syncthreads();
    float rmax[8];
    for (int g = 0; g < 8; g++) {
        int r = g + hi * 8;
        rmax[g] = fmaxf(fmaxf(Red[r], Red[16 + r]), fmaxf(Red[32 + r], Red[48 + r]));
    }
    __syncthreads();

    // Pass 2: exp + row sums
    float fs[8];
    for (int g = 0; g < 8; g++) fs[g] = 0.f;
    for (int ct = w; ct < 128; ct += 4) {
        int c = ct * 16 + n;
        for (int g = 0; g < 8; g++) {
            int m = g + hi * 8;
            float e = __expf(Sc[m * 2048 + c] - rmax[g]);
            Sc[m * 2048 + c] = e;
            fs[g] += e;
        }
    }
    for (int g = 0; g < 8; g++) fs[g] = rowsum16(fs[g]);
    if (n == 0)
        for (int g = 0; g < 8; g++) Red[w * 16 + g + hi * 8] = fs[g];
    __syncthreads();
    float rinv[8];
    for (int g = 0; g < 8; g++) {
        int r = g + hi * 8;
        rinv[g] = 1.0f / (Red[r] + Red[16 + r] + Red[32 + r] + Red[48 + r]);
    }

    // Pass 3: normalize; write fp32 attn to global, bf16 P to LDS
    for (int ct = w; ct < 128; ct += 4) {
        int c = ct * 16 + n;
        for (int g = 0; g < 8; g++) {
            int m = g + hi * 8;
            float p = Sc[m * 2048 + c] * rinv[g];
            attn_bh[(size_t)m * S_ + c] = p;
            Ps[m * 2048 + c] = (__bf16)p;
        }
    }
    __syncthreads();

    // P (16x2048) @ V (2048x64): wave w -> output cols [w*16, w*16+16)
    v8f acc = {};
    const __bf16* Bb = vbh + (size_t)(w * 16) * S_;
    for (int k0 = 0; k0 < 2048; k0 += 32)
        acc = wmma_bf16(loadA(Ps, 2048, k0), loadB(Bb + k0, 2048), acc);
    for (int g = 0; g < 8; g++) {
        int m = g + hi * 8;
        int r = qt * 16 + m;
        o[((size_t)b * S_ + r) * (H_ * DV_) + h * DV_ + w * 16 + n] = (__bf16)acc[g];
    }
}

// ---------------- Kernel 3: out = LN( o @ Wo + residual ). 256 threads = 8 waves.
__global__ __launch_bounds__(256) void out_proj_ln_k(
    const __bf16* o, const __bf16* WTo, const float* qres,
    const float* gamma, const float* beta, float* out)
{
    extern __shared__ char smem_raw[];
    __bf16* As  = (__bf16*)smem_raw;              // 16x1024 bf16 = 32 KB
    float*  Row = (float*)(smem_raw + 32768);     // 16x1024 f32  = 64 KB
    int r0 = blockIdx.x * 16;

    for (int t = threadIdx.x; t < 16 * 1024; t += 256)
        As[t] = o[(size_t)r0 * 1024 + t];
    __syncthreads();

    int w = threadIdx.x >> 5, lane = threadIdx.x & 31;
    int n = lane & 15, hi = lane >> 4;

    for (int i = 0; i < 8; i++) {
        int c0 = (w * 8 + i) * 16;
        v8f acc = {};
        const __bf16* Bb = WTo + (size_t)c0 * 1024;
        for (int k0 = 0; k0 < 1024; k0 += 32)
            acc = wmma_bf16(loadA(As, 1024, k0), loadB(Bb + k0, 1024), acc);
        for (int g = 0; g < 8; g++) {
            int m = g + hi * 8;
            int c = c0 + n;
            Row[m * 1024 + c] = acc[g] + qres[(size_t)(r0 + m) * 1024 + c];
        }
    }
    __syncthreads();

    // LayerNorm: 8 waves x 2 rows each, full-wave reductions
    for (int rr = 0; rr < 2; rr++) {
        int m = w * 2 + rr;
        float s = 0.f, s2 = 0.f;
        for (int c = lane; c < 1024; c += 32) {
            float x = Row[m * 1024 + c];
            s += x; s2 += x * x;
        }
        s = allsum32(s); s2 = allsum32(s2);
        float mu  = s * (1.0f / 1024.0f);
        float var = s2 * (1.0f / 1024.0f) - mu * mu;
        float rs  = rsqrtf(var + 1e-6f);
        for (int c = lane; c < 1024; c += 32) {
            float x = Row[m * 1024 + c];
            out[(size_t)(r0 + m) * 1024 + c] = (x - mu) * rs * gamma[c] + beta[c];
        }
    }
}

extern "C" void kernel_launch(void* const* d_in, const int* in_sizes, int n_in,
                              void* d_out, int out_size, void* d_ws, size_t ws_size,
                              hipStream_t stream)
{
    const float* q  = (const float*)d_in[0];
    const float* k  = (const float*)d_in[1];
    const float* v  = (const float*)d_in[2];
    const unsigned char* mask = (const unsigned char*)d_in[3];
    const float* Wq = (const float*)d_in[4];
    const float* Wk = (const float*)d_in[5];
    const float* Wv = (const float*)d_in[6];
    const float* Wo = (const float*)d_in[7];
    const float* gamma = (const float*)d_in[8];
    const float* beta  = (const float*)d_in[9];

    float* out  = (float*)d_out;                       // [B,S,D]
    float* attn = out + (size_t)B_ * S_ * D_;          // [B,H,S,S]

    char* ws = (char*)d_ws;
    const size_t WSZ = (size_t)1024 * 1024 * 2;        // 2 MB per bf16 weight
    const size_t HSZ = (size_t)B_ * H_ * S_ * DK_ * 2; // 16 MB per head tensor
    __bf16* WTq = (__bf16*)(ws + 0 * WSZ);
    __bf16* WTk = (__bf16*)(ws + 1 * WSZ);
    __bf16* WTv = (__bf16*)(ws + 2 * WSZ);
    __bf16* WTo = (__bf16*)(ws + 3 * WSZ);
    __bf16* qh  = (__bf16*)(ws + 4 * WSZ);
    __bf16* kh  = (__bf16*)(ws + 4 * WSZ + 1 * HSZ);
    __bf16* vT  = (__bf16*)(ws + 4 * WSZ + 2 * HSZ);
    __bf16* oB  = (__bf16*)(ws + 4 * WSZ + 3 * HSZ);

    cvt_weights_k<<<16384, 256, 0, stream>>>(Wq, Wk, Wv, Wo, WTq, WTk, WTv, WTo);

    qkv_proj_k<<<dim3((B_ * S_) / 16, 3), 256, 16 * 1024 * 2, stream>>>(
        q, k, v, WTq, WTk, WTv, qh, kh, vT);

    attention_k<<<B_ * H_ * (S_ / 16), 128, 2048 + 131072 + 65536 + 256, stream>>>(
        qh, kh, vT, mask, attn, oB);

    out_proj_ln_k<<<(B_ * S_) / 16, 256, 32768 + 65536, stream>>>(
        oB, WTo, q, gamma, beta, out);
}
